// MoETransformerBlock_12335146074166
// MI455X (gfx1250) — compile-verified, matrix operand order checked by reading, not run on hardware
//
#include <hip/hip_runtime.h>
#include <math.h>

#define B_    2
#define S_    1024
#define D_    1024
#define H_    16
#define HD_   64
#define F_    4096
#define E_    8
#define MTOK  (B_ * S_)   // 2048

typedef unsigned short u16t;
typedef __attribute__((ext_vector_type(16))) __bf16 v16bf;
typedef __attribute__((ext_vector_type(8)))  float  v8f;

// ---------------------------------------------------------------- helpers

__device__ inline u16t f2bf(float f) {
    unsigned x = __float_as_uint(f);
    unsigned r = x + 0x7fffu + ((x >> 16) & 1u);
    return (u16t)(r >> 16);
}

__device__ inline unsigned pack2(float lo, float hi) {
    return (unsigned)f2bf(lo) | ((unsigned)f2bf(hi) << 16);
}

// Load one 16x32 bf16 WMMA fragment (A layout; B uses the symmetric layout with
// the "row" meaning the B column) from memory laid out row-major with
// `stride_us` ushorts per row.  koff is in u32 units (k element offset / 2).
__device__ inline v16bf frag_ld(const u16t* base, int stride_us, int lane, int koff) {
    int r = lane & 15, hf = lane >> 4;
    const unsigned* p = (const unsigned*)(base + (long)r * stride_us);
    union { v16bf v; unsigned u[8]; } f;
#pragma unroll
    for (int q = 0; q < 8; ++q)
        f.u[q] = p[koff + (q & 3) + hf * 4 + (q >> 2) * 8];
    return f.v;
}

__device__ inline v8f wmma_bf16(v16bf a, v16bf b, v8f c) {
    return __builtin_amdgcn_wmma_f32_16x16x32_bf16(false, a, false, b, (short)0, c,
                                                   false, false);
}

__device__ inline float blk_sum(float v, float* sm) {
    int lane = threadIdx.x & 31, wv = threadIdx.x >> 5;
#pragma unroll
    for (int m = 16; m; m >>= 1) v += __shfl_xor(v, m, 32);
    __syncthreads();
    if (lane == 0) sm[wv] = v;
    __syncthreads();
    if (wv == 0) {
        float r = (lane < 8) ? sm[lane] : 0.f;
#pragma unroll
        for (int m = 4; m; m >>= 1) r += __shfl_xor(r, m, 32);
        if (lane == 0) sm[0] = r;
    }
    __syncthreads();
    return sm[0];
}

// ---------------------------------------------------------------- layernorm

__global__ __launch_bounds__(256) void ln_kernel(const float* __restrict__ x,
                                                 const float* __restrict__ g,
                                                 const float* __restrict__ bt,
                                                 float* __restrict__ y) {
    __shared__ float sm[8];
    int row = blockIdx.x, tid = threadIdx.x;
    const float4* xr = (const float4*)(x + (long)row * D_);
    float4 v = xr[tid];
    float mu = blk_sum(v.x + v.y + v.z + v.w, sm) * (1.f / D_);
    float dx = v.x - mu, dy = v.y - mu, dz = v.z - mu, dw = v.w - mu;
    float var = blk_sum(dx * dx + dy * dy + dz * dz + dw * dw, sm) * (1.f / D_);
    float rs = rsqrtf(var + 1e-3f);
    float4 gg = ((const float4*)g)[tid];
    float4 bb = ((const float4*)bt)[tid];
    float4 o;
    o.x = dx * rs * gg.x + bb.x;
    o.y = dy * rs * gg.y + bb.y;
    o.z = dz * rs * gg.z + bb.z;
    o.w = dw * rs * gg.w + bb.w;
    ((float4*)(y + (long)row * D_))[tid] = o;
}

// ---------------------------------------------------------------- WMMA GEMM
// C[M,N] = A[M,K] (f32, staged->bf16) * B[K,N] (f32, staged->bf16)
// 128x128 macro-tile, BK=32, LDS double-buffered (one barrier per k-step),
// next-tile global_prefetch.  8 waves as 4(M) x 2(N); per wave 32x64 = 8 WMMA/step.
// EPI: 0 = plain, 1 = +resid, 2 = gelu(.+bias), 3 = (.+bias)*rowscale
// Optional expert grouping via blockIdx.z / counts / rowmap.

template <int EPI>
__global__ __launch_bounds__(256) void gemm_bf16(
    const float* __restrict__ A, const float* __restrict__ Bw,
    float* __restrict__ C, int M, int N, int K,
    const float* __restrict__ resid, const float* __restrict__ bias,
    const float* __restrict__ rowscale, const int* __restrict__ rowmap,
    const int* __restrict__ counts, int rshift, long strideB, long strideBias) {
    __shared__ u16t As[2][128][40];    // [buf][m][k] bf16, padded
    __shared__ u16t Bs[2][128][36];    // [buf][n][k] bf16 (transposed), padded

    int e = blockIdx.z;
    const float* Bp    = Bw + (long)e * strideB;
    const float* biasp = bias ? bias + (long)e * strideBias : nullptr;
    const int*   mapp  = rowmap ? rowmap + e * 2048 : nullptr;
    const float* sclp  = rowscale ? rowscale + e * 2048 : nullptr;
    int cnt = counts ? counts[e] : M;

    int m0 = blockIdx.x * 128;
    int n0 = blockIdx.y * 128;
    if (m0 >= cnt) return;

    int tid = threadIdx.x;
    int lane = tid & 31, wv = tid >> 5;
    int wm = wv & 3, wn = wv >> 2;

    v8f acc[2][4];
#pragma unroll
    for (int mi = 0; mi < 2; ++mi)
#pragma unroll
        for (int ni = 0; ni < 4; ++ni)
#pragma unroll
            for (int j = 0; j < 8; ++j) acc[mi][ni][j] = 0.f;

    // A staging map: 128 rows x 32 cols, 16 f32 per thread
    int ar = tid >> 1;
    int ac = (tid & 1) * 16;
    int ai = m0 + ar;
    bool aval = ai < cnt;
    long arow = 0;
    if (aval) arow = mapp ? (long)(mapp[ai] >> rshift) : (long)ai;
    // B staging map: 32 k-rows x 128 cols, 16 f32 per thread
    int bkr = tid >> 3;
    int bc  = (tid & 7) * 16;

    float4 a0, a1, a2, a3, b0, b1, b2, b3;

#define LOAD_TILE(k0)                                                          \
    do {                                                                       \
        float4 z = {0, 0, 0, 0};                                               \
        a0 = a1 = a2 = a3 = z;                                                 \
        if (aval) {                                                            \
            const float4* ap = (const float4*)(A + arow * (long)K + (k0) + ac);\
            a0 = ap[0]; a1 = ap[1]; a2 = ap[2]; a3 = ap[3];                    \
        }                                                                      \
        const float* bp = Bp + (long)((k0) + bkr) * N + n0 + bc;               \
        b0 = ((const float4*)bp)[0]; b1 = ((const float4*)bp)[1];              \
        b2 = ((const float4*)bp)[2]; b3 = ((const float4*)bp)[3];              \
    } while (0)

#define STORE_TILE(bufi)                                                       \
    do {                                                                       \
        unsigned* asd = (unsigned*)&As[bufi][ar][ac];                          \
        asd[0] = pack2(a0.x, a0.y); asd[1] = pack2(a0.z, a0.w);                \
        asd[2] = pack2(a1.x, a1.y); asd[3] = pack2(a1.z, a1.w);                \
        asd[4] = pack2(a2.x, a2.y); asd[5] = pack2(a2.z, a2.w);                \
        asd[6] = pack2(a3.x, a3.y); asd[7] = pack2(a3.z, a3.w);                \
        float bv[16] = {b0.x, b0.y, b0.z, b0.w, b1.x, b1.y, b1.z, b1.w,        \
                        b2.x, b2.y, b2.z, b2.w, b3.x, b3.y, b3.z, b3.w};       \
        for (int i = 0; i < 16; ++i) Bs[bufi][bc + i][bkr] = f2bf(bv[i]);      \
    } while (0)

    LOAD_TILE(0);
    STORE_TILE(0);
    __syncthreads();

    int buf = 0;
    for (int k0 = 32; k0 <= K; k0 += 32) {
        if (k0 < K) {
            if (k0 + 32 < K) {   // pull the tile after next into L2/L0
                __builtin_prefetch(A + arow * (long)K + k0 + 32 + ac, 0, 0);
                __builtin_prefetch(Bp + (long)(k0 + 32 + bkr) * N + n0 + bc, 0, 0);
            }
            LOAD_TILE(k0);
            STORE_TILE(buf ^ 1);
        }
        v16bf af[2], bfv[4];
        af[0] = frag_ld(&As[buf][wm * 32][0],      40, lane, 0);
        af[1] = frag_ld(&As[buf][wm * 32 + 16][0], 40, lane, 0);
#pragma unroll
        for (int ni = 0; ni < 4; ++ni)
            bfv[ni] = frag_ld(&Bs[buf][wn * 64 + ni * 16][0], 36, lane, 0);
#pragma unroll
        for (int mi = 0; mi < 2; ++mi)
#pragma unroll
            for (int ni = 0; ni < 4; ++ni)
                acc[mi][ni] = wmma_bf16(af[mi], bfv[ni], acc[mi][ni]);
        __syncthreads();
        buf ^= 1;
    }
#undef LOAD_TILE
#undef STORE_TILE

    int cn = lane & 15, hf = lane >> 4;
#pragma unroll
    for (int mi = 0; mi < 2; ++mi)
#pragma unroll
        for (int ni = 0; ni < 4; ++ni)
#pragma unroll
            for (int j = 0; j < 8; ++j) {
                int i = m0 + wm * 32 + mi * 16 + j + 8 * hf;
                if (i >= cnt) continue;
                long crow = mapp ? (long)mapp[i] : (long)i;
                int col = n0 + wn * 64 + ni * 16 + cn;
                float v = acc[mi][ni][j];
                if (EPI == 1) v += resid[crow * (long)N + col];
                if (EPI == 2) {
                    v += biasp[col];
                    v = 0.5f * v * (1.f + erff(v * 0.70710678118f));
                }
                if (EPI == 3) v = (v + biasp[col]) * sclp[i];
                C[crow * (long)N + col] = v;
            }
}

// ---------------------------------------------------------------- RoPE / relayout
// src: [B,S, h*64+d] f32   dst: [B,H,S,HD] bf16; do_rope=0 -> plain relayout (V)

__global__ __launch_bounds__(256) void rope_kernel(const float* __restrict__ src,
                                                   u16t* __restrict__ dst, int do_rope) {
    int idx = blockIdx.x * 256 + threadIdx.x;
    int i  = idx & 31;
    int s  = (idx >> 5) & (S_ - 1);
    int bh = idx >> 15;                 // b*H + h
    int b  = bh >> 4, h = bh & 15;
    const float* row = src + ((long)(b * S_ + s)) * D_ + h * HD_;
    float x1 = row[i], x2 = row[i + 32];
    float o1 = x1, o2 = x2;
    if (do_rope) {
        float freq = expf(-9.210340372f * (float)(2 * i) * (1.f / 64.f));
        float ang  = (float)s * freq;
        float sn = sinf(ang), cs = cosf(ang);
        o1 = cs * x1 - sn * x2;
        o2 = sn * x1 + cs * x2;
    }
    u16t* drow = dst + ((long)bh * S_ + s) * HD_;
    drow[i]      = f2bf(o1);
    drow[i + 32] = f2bf(o2);
}

// ---------------------------------------------------------------- flash attention
// One block = 128 q rows of one (b,h); 8 waves, 16 q rows each; 32-key tiles.

__global__ __launch_bounds__(256) void flash_kernel(const u16t* __restrict__ qr,
                                                    const u16t* __restrict__ kr,
                                                    const u16t* __restrict__ vr,
                                                    float* __restrict__ ctx) {
    __shared__ u16t Ks[32][72];       // [key][d]
    __shared__ u16t Vt[64][40];       // [d][key]  (transposed)
    __shared__ u16t Ps[8][16][40];    // per-wave P tile [m][key]

    int b = blockIdx.z, h = blockIdx.y;
    int q0 = blockIdx.x * 128;
    int tid = threadIdx.x, lane = tid & 31, w = tid >> 5;
    int qbase = q0 + w * 16;
    long bh = (long)(b * H_ + h);

    const u16t* qp = qr + (bh * S_ + qbase) * HD_;
    v16bf qf[2];
    qf[0] = frag_ld(qp, HD_, lane, 0);
    qf[1] = frag_ld(qp, HD_, lane, 16);

    v8f O[4];
#pragma unroll
    for (int d = 0; d < 4; ++d)
#pragma unroll
        for (int j = 0; j < 8; ++j) O[d][j] = 0.f;
    float mrow[8], lrow[8];
#pragma unroll
    for (int j = 0; j < 8; ++j) { mrow[j] = -3e38f; lrow[j] = 0.f; }

    int cn = lane & 15, hf = lane >> 4;
    int kend = q0 + 128;

    for (int kt = 0; kt < kend; kt += 32) {
        __syncthreads();
        {   // stage K (row-major) and V (transposed) tiles
            int key = tid >> 3, ch = tid & 7;
            if (kt + 32 < kend) {   // prefetch next key tile
                __builtin_prefetch(kr + (bh * S_ + kt + 32 + key) * HD_ + ch * 8, 0, 0);
                __builtin_prefetch(vr + (bh * S_ + kt + 32 + key) * HD_ + ch * 8, 0, 0);
            }
            const uint4* ksrc = (const uint4*)(kr + (bh * S_ + kt + key) * HD_ + ch * 8);
            *((uint4*)&Ks[key][ch * 8]) = *ksrc;
            u16t tmp[8];
            *(uint4*)tmp = *(const uint4*)(vr + (bh * S_ + kt + key) * HD_ + ch * 8);
#pragma unroll
            for (int i = 0; i < 8; ++i) Vt[ch * 8 + i][key] = tmp[i];
        }
        __syncthreads();

        // scores: 16 q rows x 32 keys
        v8f sc[2];
#pragma unroll
        for (int ni = 0; ni < 2; ++ni) {
            v8f s;
#pragma unroll
            for (int j = 0; j < 8; ++j) s[j] = 0.f;
            v16bf bk0 = frag_ld(&Ks[ni * 16][0], 72, lane, 0);
            v16bf bk1 = frag_ld(&Ks[ni * 16][0], 72, lane, 16);
            s = wmma_bf16(qf[0], bk0, s);
            s = wmma_bf16(qf[1], bk1, s);
            sc[ni] = s;
        }

        float nm[8], alpha[8];
#pragma unroll
        for (int j = 0; j < 8; ++j) {
            int row = qbase + j + 8 * hf;
#pragma unroll
            for (int ni = 0; ni < 2; ++ni) {
                int col = kt + ni * 16 + cn;
                if (col > row) sc[ni][j] = -1e9f;
                else           sc[ni][j] *= 0.125f;   // 1/sqrt(64)
            }
            float mx = fmaxf(sc[0][j], sc[1][j]);
#pragma unroll
            for (int msk = 8; msk; msk >>= 1) mx = fmaxf(mx, __shfl_xor(mx, msk, 16));
            nm[j] = fmaxf(mrow[j], mx);
        }
#pragma unroll
        for (int j = 0; j < 8; ++j) {
            alpha[j] = expf(mrow[j] - nm[j]);
            mrow[j] = nm[j];
            float p0 = expf(sc[0][j] - nm[j]);
            float p1 = expf(sc[1][j] - nm[j]);
            sc[0][j] = p0; sc[1][j] = p1;
            float rs = p0 + p1;
#pragma unroll
            for (int msk = 8; msk; msk >>= 1) rs += __shfl_xor(rs, msk, 16);
            lrow[j] = lrow[j] * alpha[j] + rs;
        }
#pragma unroll
        for (int d = 0; d < 4; ++d)
#pragma unroll
            for (int j = 0; j < 8; ++j) O[d][j] *= alpha[j];

        // route P through LDS into A-fragment layout
#pragma unroll
        for (int ni = 0; ni < 2; ++ni)
#pragma unroll
            for (int j = 0; j < 8; ++j)
                Ps[w][j + 8 * hf][ni * 16 + cn] = f2bf(sc[ni][j]);
        __syncthreads();

        v16bf pf = frag_ld(&Ps[w][0][0], 40, lane, 0);
#pragma unroll
        for (int d = 0; d < 4; ++d) {
            v16bf vf = frag_ld(&Vt[d * 16][0], 40, lane, 0);
            O[d] = wmma_bf16(pf, vf, O[d]);
        }
    }

#pragma unroll
    for (int j = 0; j < 8; ++j) lrow[j] = 1.f / lrow[j];
#pragma unroll
    for (int d = 0; d < 4; ++d)
#pragma unroll
        for (int j = 0; j < 8; ++j) {
            int row = qbase + j + 8 * hf;
            int col = h * HD_ + d * 16 + cn;
            ctx[((long)(b * S_ + row)) * D_ + col] = O[d][j] * lrow[j];
        }
}

// ---------------------------------------------------------------- router & routing

__global__ __launch_bounds__(256) void router_kernel(const float* __restrict__ h2,
                                                     const float* __restrict__ wr,
                                                     float* __restrict__ logits) {
    int gid = blockIdx.x * 256 + threadIdx.x;   // 2048*8 pairs
    int t = gid >> 3, e = gid & 7;
    const float* hp = h2 + (long)t * D_;
    float acc = 0.f;
    for (int d = 0; d < D_; ++d) acc += hp[d] * wr[d * E_ + e];
    logits[t * E_ + e] = acc;
}

__global__ __launch_bounds__(256) void zero_counts(int* cnt) {
    if (threadIdx.x < E_) cnt[threadIdx.x] = 0;
}

__global__ __launch_bounds__(256) void route_kernel(const float* __restrict__ logits,
                                                    float* __restrict__ probs,
                                                    float* __restrict__ lse,
                                                    int* __restrict__ cnt,
                                                    int* __restrict__ rowmap,
                                                    float* __restrict__ wl) {
    int t = blockIdx.x * 256 + threadIdx.x;
    if (t >= MTOK) return;
    float lg[E_], p[E_];
    float m = -3e38f;
#pragma unroll
    for (int e = 0; e < E_; ++e) { lg[e] = logits[t * E_ + e]; m = fmaxf(m, lg[e]); }
    float s = 0.f;
#pragma unroll
    for (int e = 0; e < E_; ++e) { p[e] = expf(lg[e] - m); s += p[e]; }
    float inv = 1.f / s;
#pragma unroll
    for (int e = 0; e < E_; ++e) { p[e] *= inv; probs[t * E_ + e] = p[e]; }
    lse[t] = m + logf(s);
    int i0 = 0;
#pragma unroll
    for (int e = 1; e < E_; ++e) if (p[e] > p[i0]) i0 = e;
    int i1 = (i0 == 0) ? 1 : 0;
#pragma unroll
    for (int e = 0; e < E_; ++e) if (e != i0 && p[e] > p[i1]) i1 = e;
    float den = p[i0] + p[i1] + 1e-8f;
    int pos = atomicAdd(&cnt[i0], 1);
    rowmap[i0 * 2048 + pos] = t * 2;
    wl[i0 * 2048 + pos]     = p[i0] / den;
    pos = atomicAdd(&cnt[i1], 1);
    rowmap[i1 * 2048 + pos] = t * 2 + 1;
    wl[i1 * 2048 + pos]     = p[i1] / den;
}

__global__ __launch_bounds__(256) void loss_kernel(const float* __restrict__ probs,
                                                   const float* __restrict__ lse,
                                                   float* __restrict__ lb_out,
                                                   float* __restrict__ zl_out) {
    __shared__ float sm[8];
    float us[E_];
#pragma unroll
    for (int e = 0; e < E_; ++e) us[e] = 0.f;
    float zs = 0.f;
    for (int t = threadIdx.x; t < MTOK; t += 256) {
#pragma unroll
        for (int e = 0; e < E_; ++e) us[e] += probs[t * E_ + e];
        float L = lse[t];
        zs += L * L;
    }
    float ur[E_];
    for (int e = 0; e < E_; ++e) ur[e] = blk_sum(us[e], sm);
    float z = blk_sum(zs, sm);
    if (threadIdx.x == 0) {
        float usage[E_], mean = 0.f;
        for (int e = 0; e < E_; ++e) { usage[e] = ur[e] * (1.f / MTOK); mean += usage[e]; }
        mean *= (1.f / E_);
        float var = 0.f;
        for (int e = 0; e < E_; ++e) { float d = usage[e] - mean; var += d * d; }
        var *= (1.f / E_);
        *lb_out = var / (mean * mean + 1e-8f) * (float)E_ * 0.01f;
        *zl_out = z * (1.f / MTOK) * 0.001f;
    }
}

// ---------------------------------------------------------------- final combine

__global__ __launch_bounds__(256) void combine_kernel(const float* __restrict__ x2,
                                                      const float* __restrict__ part,
                                                      float* __restrict__ out) {
    int t = blockIdx.x, c = threadIdx.x;
    float4 a  = ((const float4*)(x2   + (long)t * D_))[c];
    float4 p0 = ((const float4*)(part + (long)(2 * t) * D_))[c];
    float4 p1 = ((const float4*)(part + (long)(2 * t + 1) * D_))[c];
    float4 r;
    r.x = a.x + p0.x + p1.x;
    r.y = a.y + p0.y + p1.y;
    r.z = a.z + p0.z + p1.z;
    r.w = a.w + p0.w + p1.w;
    ((float4*)(out + (long)t * D_))[c] = r;
}

// ---------------------------------------------------------------- launch

extern "C" void kernel_launch(void* const* d_in, const int* in_sizes, int n_in,
                              void* d_out, int out_size, void* d_ws, size_t ws_size,
                              hipStream_t stream) {
    (void)in_sizes; (void)n_in; (void)out_size; (void)ws_size;
    const float* x        = (const float*)d_in[0];
    const float* ln1_g    = (const float*)d_in[1];
    const float* ln1_b    = (const float*)d_in[2];
    const float* wq       = (const float*)d_in[3];
    const float* wk       = (const float*)d_in[4];
    const float* wv       = (const float*)d_in[5];
    const float* wo       = (const float*)d_in[6];
    const float* ln2_g    = (const float*)d_in[7];
    const float* ln2_b    = (const float*)d_in[8];
    const float* w_router = (const float*)d_in[9];
    const float* w1       = (const float*)d_in[10];
    const float* b1       = (const float*)d_in[11];
    const float* w2       = (const float*)d_in[12];
    const float* b2       = (const float*)d_in[13];
    float* out = (float*)d_out;

    char* wp = (char*)d_ws;
    auto alloc = [&](size_t bytes) {
        void* p = (void*)wp;
        wp += (bytes + 255) & ~(size_t)255;
        return p;
    };
    float* h1    = (float*)alloc((size_t)MTOK * D_ * 4);
    float* qf    = (float*)alloc((size_t)MTOK * D_ * 4);
    float* kf    = (float*)alloc((size_t)MTOK * D_ * 4);
    float* vf    = (float*)alloc((size_t)MTOK * D_ * 4);
    u16t*  qrb   = (u16t*)alloc((size_t)MTOK * D_ * 2);
    u16t*  krb   = (u16t*)alloc((size_t)MTOK * D_ * 2);
    u16t*  vrb   = (u16t*)alloc((size_t)MTOK * D_ * 2);
    float* ctx   = (float*)alloc((size_t)MTOK * D_ * 4);
    float* x2v   = (float*)alloc((size_t)MTOK * D_ * 4);
    float* h2    = (float*)alloc((size_t)MTOK * D_ * 4);
    float* logit = (float*)alloc((size_t)MTOK * E_ * 4);
    float* probs = (float*)alloc((size_t)MTOK * E_ * 4);
    float* lse   = (float*)alloc((size_t)MTOK * 4);
    int*   cnt   = (int*)alloc(256);
    int*   rowmap= (int*)alloc((size_t)E_ * 2048 * 4);
    float* wl    = (float*)alloc((size_t)E_ * 2048 * 4);
    float* hhb   = (float*)alloc((size_t)2 * MTOK * F_ * 4);   // [token*2+k][F]
    float* part  = (float*)alloc((size_t)2 * MTOK * D_ * 4);   // [token*2+k][D]

    dim3 blk(256);
    zero_counts<<<1, 32, 0, stream>>>(cnt);

    ln_kernel<<<MTOK, blk, 0, stream>>>(x, ln1_g, ln1_b, h1);

    dim3 gproj(MTOK / 128, D_ / 128, 1);
    gemm_bf16<0><<<gproj, blk, 0, stream>>>(h1, wq, qf, MTOK, D_, D_,
        nullptr, nullptr, nullptr, nullptr, nullptr, 0, 0, 0);
    gemm_bf16<0><<<gproj, blk, 0, stream>>>(h1, wk, kf, MTOK, D_, D_,
        nullptr, nullptr, nullptr, nullptr, nullptr, 0, 0, 0);
    gemm_bf16<0><<<gproj, blk, 0, stream>>>(h1, wv, vf, MTOK, D_, D_,
        nullptr, nullptr, nullptr, nullptr, nullptr, 0, 0, 0);

    int rope_grid = (B_ * H_ * S_ * 32) / 256;
    rope_kernel<<<rope_grid, blk, 0, stream>>>(qf, qrb, 1);
    rope_kernel<<<rope_grid, blk, 0, stream>>>(kf, krb, 1);
    rope_kernel<<<rope_grid, blk, 0, stream>>>(vf, vrb, 0);

    dim3 gattn(S_ / 128, H_, B_);
    flash_kernel<<<gattn, blk, 0, stream>>>(qrb, krb, vrb, ctx);

    gemm_bf16<1><<<gproj, blk, 0, stream>>>(ctx, wo, x2v, MTOK, D_, D_,
        x, nullptr, nullptr, nullptr, nullptr, 0, 0, 0);

    ln_kernel<<<MTOK, blk, 0, stream>>>(x2v, ln2_g, ln2_b, h2);

    router_kernel<<<(MTOK * E_) / 256, blk, 0, stream>>>(h2, w_router, logit);
    route_kernel<<<MTOK / 256, blk, 0, stream>>>(logit, probs, lse, cnt, rowmap, wl);
    loss_kernel<<<1, blk, 0, stream>>>(probs, lse,
        out + (size_t)MTOK * D_, out + (size_t)MTOK * D_ + 1);

    dim3 gexp1(MTOK / 128, F_ / 128, E_);
    gemm_bf16<2><<<gexp1, blk, 0, stream>>>(h2, w1, hhb, MTOK, F_, D_,
        nullptr, b1, nullptr, rowmap, cnt, 1, (long)D_ * F_, (long)F_);
    dim3 gexp2(MTOK / 128, D_ / 128, E_);
    gemm_bf16<3><<<gexp2, blk, 0, stream>>>(hhb, w2, part, MTOK, D_, F_,
        nullptr, b2, wl, rowmap, cnt, 0, (long)F_ * D_, (long)D_);

    combine_kernel<<<MTOK, blk, 0, stream>>>(x2v, part, out);
}
